// GRULayer_47218870452491
// MI455X (gfx1250) — matrix operand order repaired and checked by reading, not compiled, not run on hardware
//
#include <hip/hip_runtime.h>
#include <hip/hip_bf16.h>

// ---------------------------------------------------------------------------
// GRU layer for MI455X (gfx1250, wave32).
//   T=256, B=128, D=1024, H=1024
// Phase 1: gi = x @ w_ih^T + b_ih  (bf16 WMMA, f32 accum), A staged to LDS by
//          the Tensor Data Mover (double-buffered 128x32 chunks).
// Phase 2: 256 sequential fused step kernels: TDM stages the 16x1024 h tile
//          once per block; gh = h @ w_hh^T ; gate math ; h_t -> d_out + bf16.
// ---------------------------------------------------------------------------

typedef __attribute__((ext_vector_type(4)))  __bf16 v4bf;
typedef __attribute__((ext_vector_type(8)))  __bf16 v8bf;
typedef __attribute__((ext_vector_type(16))) __bf16 v16bf;
typedef __attribute__((ext_vector_type(8)))  float  v8f;

typedef __attribute__((ext_vector_type(4))) unsigned int tdm_g0_t;
typedef __attribute__((ext_vector_type(8))) int          tdm_g1_t;
typedef __attribute__((ext_vector_type(4))) int          tdm_g2_t;

#define GRU_T 256
#define GRU_B 128
#define GRU_D 1024
#define GRU_H 1024
#define GRU_3H 3072

#if defined(__gfx1250__) && __has_builtin(__builtin_amdgcn_tensor_load_to_lds) && \
    __has_builtin(__builtin_amdgcn_s_wait_tensorcnt)
#define GRU_HAVE_TDM 1
#else
#define GRU_HAVE_TDM 0
#endif

// ---- WMMA helper ----------------------------------------------------------

__device__ __forceinline__ v8f wmma_bf16(v16bf a, v16bf b, v8f c) {
  return __builtin_amdgcn_wmma_f32_16x16x32_bf16(
      /*neg_a=*/false, a, /*neg_b=*/false, b,
      /*c_mod=*/(short)0, c, /*reuse_a=*/false, /*reuse_b=*/false);
}

// B fragment: 32x16 bf16 tile. Source weight layout [N, K] row-major.
// Lane l (l<16): column n=l, K=[0..7,16..23]; lane l>=16: column n=l-16, K=[8..15,24..31]
__device__ __forceinline__ v16bf load_frag_b(const __bf16* __restrict__ base,
                                             int ldb, int col0, int k0, int lane) {
  const int n    = lane & 15;
  const int half = lane >> 4;
  const __bf16* p = base + (size_t)(col0 + n) * ldb + (k0 + half * 8);
  union { v16bf v; v8bf h[2]; } u;
  u.h[0] = *(const v8bf*)p;
  u.h[1] = *(const v8bf*)(p + 16);
  return u.v;
}

// ---- Tensor Data Mover: 2D tile load (D# per CDNA5 ISA) -------------

#if GRU_HAVE_TDM
__device__ __forceinline__ void tdm_load_2d(unsigned lds_addr,
                                            unsigned long long gaddr,
                                            tdm_g1_t g1) {
  tdm_g0_t g0;
  g0[0] = 1u;                                   // count=1, user mode, no gather
  g0[1] = lds_addr;                             // D#.lds_addr
  g0[2] = (unsigned)(gaddr & 0xFFFFFFFFull);    // D#.global_addr[31:0]
  g0[3] = (unsigned)((gaddr >> 32) & 0x01FFFFFFull) | (2u << 30);  // [56:32] | type=2
  tdm_g2_t z4 = {0, 0, 0, 0};
#if defined(__clang_major__) && (__clang_major__ >= 23)
  tdm_g1_t z8 = {0, 0, 0, 0, 0, 0, 0, 0};
  __builtin_amdgcn_tensor_load_to_lds(g0, g1, z4, z4, z8, 0);
#else
  __builtin_amdgcn_tensor_load_to_lds(g0, g1, z4, z4, 0);
#endif
}
#endif

// ---- conversion kernel ----------------------------------------------------

__global__ __launch_bounds__(256) void f32_to_bf16_kernel(
    const float* __restrict__ in, __bf16* __restrict__ out, size_t n4) {
  size_t i = ((size_t)blockIdx.x * blockDim.x + threadIdx.x);
  if (i >= n4) return;
  size_t idx = i * 4;
  float4 f = *(const float4*)(in + idx);
  v4bf o;
  o[0] = (__bf16)f.x; o[1] = (__bf16)f.y; o[2] = (__bf16)f.z; o[3] = (__bf16)f.w;
  *(v4bf*)(out + idx) = o;
}

// ---- Phase 1: gi = x @ w_ih^T + b_ih --------------------------------------
// C[32768,3072]. Block = 8 waves -> 128x128 macro tile; wave = 64x32.
// A chunk 128 rows x 32 K staged to LDS by TDM, double-buffered.
// LDS layout: row stride 80B (64B data + 16B TDM pad) -> conflict-free b128.

#define GEMM_AROW_STRIDE 80
#define GEMM_ABUF_BYTES  (128 * GEMM_AROW_STRIDE)  // 10240

__global__ __launch_bounds__(256) void gru_gemm_gi(
    const __bf16* __restrict__ A,      // [32768,1024] bf16
    const __bf16* __restrict__ W,      // [3072,1024] bf16
    const float*  __restrict__ bias,   // [3072]
    float* __restrict__ C) {           // [32768,3072]
  __shared__ __attribute__((aligned(16))) unsigned char sA[2][GEMM_ABUF_BYTES];

  const int lane = threadIdx.x & 31;
  const int wave = threadIdx.x >> 5;          // 0..7
  const int wr   = wave >> 2;                 // 0..1: 64-row half
  const int rblk = blockIdx.y * 128;
  const int c0   = blockIdx.x * 128 + (wave & 3) * 32;

#if GRU_HAVE_TDM
  // D# group1: data_size=2B, pad every 64B by 16B, tensor 1024x32768,
  // tile 32(K) x 128(rows), dim0 stride 1024 elements.
  const tdm_g1_t g1 = {
      (int)((1u << 16) | (1u << 20) | (3u << 22) | (3u << 25)),
      (int)(1024u << 16),          // tensor_dim0 = 1024
      (int)(32768u << 16),         // tensor_dim1 = 32768
      (int)(32u << 16),            // tile_dim0 = 32
      (int)128,                    // tile_dim1 = 128
      (int)1024,                   // tensor_dim0_stride = 1024
      0, 0};
  const bool issuer =
      ((__builtin_amdgcn_readfirstlane((int)threadIdx.x) >> 5) == 0);
  const unsigned long long Abase =
      (unsigned long long)(uintptr_t)(A + (size_t)rblk * GRU_D);
  if (issuer) {
    tdm_load_2d((unsigned)(uintptr_t)&sA[0][0], Abase, g1);
  }
#endif

  const v8f vzero = {0.f, 0.f, 0.f, 0.f, 0.f, 0.f, 0.f, 0.f};
  v8f acc[4][2];
#pragma unroll
  for (int i = 0; i < 4; ++i)
#pragma unroll
    for (int j = 0; j < 2; ++j) acc[i][j] = vzero;

  for (int kc = 0; kc < 32; ++kc) {
    const int k0 = kc * 32;
#if GRU_HAVE_TDM
    if (issuer) {
      if (kc + 1 < 32) {
        tdm_load_2d((unsigned)(uintptr_t)&sA[(kc + 1) & 1][0],
                    Abase + (size_t)(k0 + 32) * 2, g1);
        __builtin_amdgcn_s_wait_tensorcnt(1);   // current chunk done
      } else {
        __builtin_amdgcn_s_wait_tensorcnt(0);
      }
    }
    __syncthreads();
#else
    // Fallback: cooperative staged copy into padded layout
    for (int c = threadIdx.x; c < 512; c += 256) {
      const int row = c >> 2, off = (c & 3) * 16;
      *(v8bf*)(&sA[kc & 1][0] + row * GEMM_AROW_STRIDE + off) =
          *(const v8bf*)((const unsigned char*)A +
                         (size_t)(rblk + row) * (GRU_D * 2) + (size_t)k0 * 2 + off);
    }
    __syncthreads();
#endif
    const unsigned char* sb = &sA[kc & 1][0];
    const int m    = lane & 15;
    const int half = lane >> 4;

    v16bf a[4], b[2];
#pragma unroll
    for (int i = 0; i < 4; ++i) {
      const unsigned char* p =
          sb + (size_t)(wr * 64 + i * 16 + m) * GEMM_AROW_STRIDE + half * 16;
      union { v16bf v; v8bf h[2]; } u;
      u.h[0] = *(const v8bf*)p;          // K = half*8 .. +7
      u.h[1] = *(const v8bf*)(p + 32);   // K = half*8+16 .. +7
      a[i] = u.v;
    }
    if (k0 + 64 < GRU_D)
      __builtin_prefetch(W + (size_t)(c0 + m) * GRU_D + k0 + 64, 0, 0);
#pragma unroll
    for (int j = 0; j < 2; ++j) b[j] = load_frag_b(W, GRU_D, c0 + 16 * j, k0, lane);
#pragma unroll
    for (int i = 0; i < 4; ++i)
#pragma unroll
      for (int j = 0; j < 2; ++j) acc[i][j] = wmma_bf16(a[i], b[j], acc[i][j]);
    __syncthreads();   // all reads done before next TDM overwrites this buffer
  }

  const int nn = lane & 15;
  const int rh = (lane >> 4) * 8;
#pragma unroll
  for (int j = 0; j < 2; ++j) {
    const int col = c0 + 16 * j + nn;
    const float bv = bias[col];
#pragma unroll
    for (int i = 0; i < 4; ++i) {
#pragma unroll
      for (int v = 0; v < 8; ++v) {
        const int row = rblk + wr * 64 + 16 * i + rh + v;
        C[(size_t)row * GRU_3H + col] = acc[i][j][v] + bv;
      }
    }
  }
}

// ---- Phase 2: fused recurrent step ----------------------------------------
// 64 blocks x 8 waves; block = 16 batch rows x (8 x 16) hidden cols.
// TDM stages the full 16x1024 h tile once (as a 32x512 tile, 16B pad per
// 1024B half-row -> 1040B half-row stride), then 32 K-iterations read A
// fragments from LDS while 3 gate weight streams come from the L2-resident
// w_hh. Epilogue fuses the sigmoid/tanh gating.

#define STEP_HALF_STRIDE 1040
#define STEP_ABUF_BYTES  (32 * STEP_HALF_STRIDE)  // 33280

__global__ __launch_bounds__(256) void gru_step(
    const __bf16* __restrict__ hbf_in,   // [B,H] bf16 h_{t-1}
    const float*  __restrict__ hprev,    // [B,H] f32  h_{t-1}
    const __bf16* __restrict__ Whh,      // [3H,H] bf16
    const float*  __restrict__ bhh,      // [3H]
    const float*  __restrict__ gi,       // [B,3H] f32 (b_ih already added)
    float*        __restrict__ hout,     // [B,H] f32  h_t (-> outputs[t])
    __bf16*       __restrict__ hbf_out)  // [B,H] bf16 h_t
{
  __shared__ __attribute__((aligned(16))) unsigned char sA[STEP_ABUF_BYTES];

  const int lane = threadIdx.x & 31;
  const int wid  = blockIdx.x * 8 + (threadIdx.x >> 5);  // 0..511
  const int r0   = (wid >> 6) * 16;   // batch-row tile (block-uniform)
  const int c0   = (wid & 63) * 16;   // hidden-col tile

#if GRU_HAVE_TDM
  if ((__builtin_amdgcn_readfirstlane((int)threadIdx.x) >> 5) == 0) {
    const int r0s = (blockIdx.x >> 3) * 16;
    // D# group1: data_size=2B, pad every 1024B by 16B, tile 512 x 32.
    const tdm_g1_t g1 = {
        (int)((1u << 16) | (1u << 20) | (7u << 22) | (3u << 25)),
        (int)(512u << 16),           // tensor_dim0 = 512
        (int)(32u << 16),            // tensor_dim1 = 32
        (int)(512u << 16),           // tile_dim0 = 512
        (int)32,                     // tile_dim1 = 32
        (int)512,                    // tensor_dim0_stride = 512
        0, 0};
    tdm_load_2d((unsigned)(uintptr_t)&sA[0],
                (unsigned long long)(uintptr_t)(hbf_in + (size_t)r0s * GRU_H), g1);
    __builtin_amdgcn_s_wait_tensorcnt(0);
  }
  __syncthreads();
#else
  for (int c = threadIdx.x; c < 2048; c += 256) {   // 2048 x 16B chunks
    const int half = c >> 6, off = (c & 63) * 16;
    *(v8bf*)(&sA[0] + half * STEP_HALF_STRIDE + off) =
        *(const v8bf*)((const unsigned char*)hbf_in + (size_t)r0 * (GRU_H * 2) +
                       (size_t)c * 16);
  }
  __syncthreads();
#endif

  const v8f vzero = {0.f, 0.f, 0.f, 0.f, 0.f, 0.f, 0.f, 0.f};
  v8f aR = vzero, aZ = vzero, aN = vzero;

  const int m    = lane & 15;
  const int half = lane >> 4;

  for (int k0 = 0; k0 < GRU_H; k0 += 32) {
    const int k = k0 + half * 8;
    const unsigned char* p =
        &sA[0] + (size_t)m * (2 * STEP_HALF_STRIDE) +
        (size_t)(k >> 9) * STEP_HALF_STRIDE + (size_t)(k & 511) * 2;
    union { v16bf v; v8bf h[2]; } u;
    u.h[0] = *(const v8bf*)p;          // K = k .. k+7
    u.h[1] = *(const v8bf*)(p + 32);   // K = k+16 .. k+23
    const v16bf a = u.v;

    const v16bf bR = load_frag_b(Whh, GRU_H, c0,             k0, lane);
    const v16bf bZ = load_frag_b(Whh, GRU_H, GRU_H + c0,     k0, lane);
    const v16bf bN = load_frag_b(Whh, GRU_H, 2 * GRU_H + c0, k0, lane);
    aR = wmma_bf16(a, bR, aR);
    aZ = wmma_bf16(a, bZ, aZ);
    aN = wmma_bf16(a, bN, aN);
  }

  const int nn  = lane & 15;
  const int rh  = (lane >> 4) * 8;
  const int col = c0 + nn;
  const float bR = bhh[col];
  const float bZ = bhh[GRU_H + col];
  const float bN = bhh[2 * GRU_H + col];
#pragma unroll
  for (int v = 0; v < 8; ++v) {
    const int row = r0 + rh + v;
    const size_t gib = (size_t)row * GRU_3H;
    const float gr = gi[gib + col]         + aR[v] + bR;
    const float gz = gi[gib + GRU_H + col] + aZ[v] + bZ;
    const float gn = gi[gib + 2 * GRU_H + col];
    const float r = 1.f / (1.f + __expf(-gr));
    const float z = 1.f / (1.f + __expf(-gz));
    const float n = tanhf(gn + r * (aN[v] + bN));
    const float hp = hprev[(size_t)row * GRU_H + col];
    const float h  = (1.f - z) * n + z * hp;
    hout[(size_t)row * GRU_H + col]    = h;
    hbf_out[(size_t)row * GRU_H + col] = (__bf16)h;
  }
}

// ---------------------------------------------------------------------------

extern "C" void kernel_launch(void* const* d_in, const int* in_sizes, int n_in,
                              void* d_out, int out_size, void* d_ws, size_t ws_size,
                              hipStream_t stream) {
  const float* x   = (const float*)d_in[0];  // [T,B,D]
  const float* st  = (const float*)d_in[1];  // [B,H]
  const float* wih = (const float*)d_in[2];  // [3H,D]
  const float* whh = (const float*)d_in[3];  // [3H,H]
  const float* bih = (const float*)d_in[4];  // [3H]
  const float* bhh = (const float*)d_in[5];  // [3H]
  float* out = (float*)d_out;                // [T,B,H] ++ [B,H]

  const size_t M = (size_t)GRU_T * GRU_B;    // 32768

  char* ws = (char*)d_ws;
  __bf16* x_bf   = (__bf16*)ws; ws += M * GRU_D * sizeof(__bf16);
  __bf16* wih_bf = (__bf16*)ws; ws += (size_t)GRU_3H * GRU_D * sizeof(__bf16);
  __bf16* whh_bf = (__bf16*)ws; ws += (size_t)GRU_3H * GRU_H * sizeof(__bf16);
  __bf16* hbf0   = (__bf16*)ws; ws += (size_t)GRU_B * GRU_H * sizeof(__bf16);
  __bf16* hbf1   = (__bf16*)ws; ws += (size_t)GRU_B * GRU_H * sizeof(__bf16);
  float*  gi     = (float*)ws;  // [T,B,3H] f32

  {
    size_t n4;
    n4 = M * GRU_D / 4;
    f32_to_bf16_kernel<<<(unsigned)((n4 + 255) / 256), 256, 0, stream>>>(x, x_bf, n4);
    n4 = (size_t)GRU_3H * GRU_D / 4;
    f32_to_bf16_kernel<<<(unsigned)((n4 + 255) / 256), 256, 0, stream>>>(wih, wih_bf, n4);
    n4 = (size_t)GRU_3H * GRU_H / 4;
    f32_to_bf16_kernel<<<(unsigned)((n4 + 255) / 256), 256, 0, stream>>>(whh, whh_bf, n4);
    n4 = (size_t)GRU_B * GRU_H / 4;
    f32_to_bf16_kernel<<<(unsigned)((n4 + 255) / 256), 256, 0, stream>>>(st, hbf0, n4);
  }

  {
    dim3 grid(GRU_3H / 128, (unsigned)(M / 128));  // 24 x 256
    gru_gemm_gi<<<grid, 256, 0, stream>>>(x_bf, wih_bf, bih, gi);
  }

  for (int t = 0; t < GRU_T; ++t) {
    const __bf16* hin  = (t & 1) ? hbf1 : hbf0;
    __bf16*       hnxt = (t & 1) ? hbf0 : hbf1;
    const float*  hpre = (t == 0) ? st : (out + (size_t)(t - 1) * GRU_B * GRU_H);
    gru_step<<<64, 256, 0, stream>>>(
        hin, hpre, whh_bf, bhh,
        gi + (size_t)t * GRU_B * GRU_3H,
        out + (size_t)t * GRU_B * GRU_H,
        hnxt);
  }

  hipMemcpyAsync(out + (size_t)GRU_T * GRU_B * GRU_H,
                 out + (size_t)(GRU_T - 1) * GRU_B * GRU_H,
                 (size_t)GRU_B * GRU_H * sizeof(float),
                 hipMemcpyDeviceToDevice, stream);
}
